// DeepFM_45208825757783
// MI455X (gfx1250) — matrix operand order repaired and checked by reading, not compiled, not run on hardware
//
#include <hip/hip_runtime.h>

// ---------------------------------------------------------------------------
// DeepFM forward for MI455X (gfx1250): bf16 WMMA for the MLP GEMMs,
// wave32 gather/reduce kernels for the FM + embedding part.
// Round 5: fully-unrolled GEMM k-loops + __builtin_amdgcn_sched_group_barrier
// pipeline spec (VMEM-read groups one k-step ahead of WMMA groups) so the
// post-RA scheduler keeps loads in flight across the matrix ops instead of
// draining loadcnt to 0 before every WMMA.
// ---------------------------------------------------------------------------

constexpr int kB   = 16384;
constexpr int kF   = 26;
constexpr int kV   = 100000;
constexpr int kE   = 32;
constexpr int kL   = 20;
constexpr int kDVF = 128;
constexpr int kH1  = 512;
constexpr int kH2  = 256;
constexpr int kK1  = (kF + 2) * kE;   // 896 = deep_emb width

typedef __bf16 bf16x8  __attribute__((ext_vector_type(8)));
typedef __bf16 bf16x16 __attribute__((ext_vector_type(16)));
typedef float  f32x8   __attribute__((ext_vector_type(8)));

#if defined(__has_builtin)
#if __has_builtin(__builtin_amdgcn_sched_group_barrier)
#define SCHED_GROUP(mask, size, id) __builtin_amdgcn_sched_group_barrier(mask, size, id)
#endif
#endif
#ifndef SCHED_GROUP
#define SCHED_GROUP(mask, size, id)
#endif
// mask values (LLVM AMDGPU): 0x0020 = VMEM read, 0x0008 = MFMA/WMMA
#define SG_VMEM_READ 0x0020
#define SG_WMMA      0x0008

__device__ __forceinline__ __bf16 f2bf(float f) {
    unsigned u = __builtin_bit_cast(unsigned, f);
    u += 0x7FFFu + ((u >> 16) & 1u);                 // round-to-nearest-even
    return __builtin_bit_cast(__bf16, (unsigned short)(u >> 16));
}

// ---------------------------------------------------------------------------
// Kernel 0: W1 (896x512 f32, K-major) -> W1T (512x896 bf16, N-major)
//           W2 (512x256 f32)          -> W2T (256x512 bf16)
// so WMMA B-fragments are two contiguous 16B chunks per lane.
// ---------------------------------------------------------------------------
__global__ __launch_bounds__(256) void deepfm_prep(
    const float* __restrict__ W1, const float* __restrict__ W2,
    __bf16* __restrict__ W1T, __bf16* __restrict__ W2T)
{
    int id = blockIdx.x * blockDim.x + threadIdx.x;
    const int N1 = kK1 * kH1;      // 458752
    const int N2 = kH1 * kH2;      // 131072
    if (id < N1) {
        int k = id / kH1, n = id % kH1;
        W1T[(size_t)n * kK1 + k] = f2bf(W1[id]);
    } else if (id < N1 + N2) {
        int i = id - N1;
        int k = i / kH2, n = i % kH2;
        W2T[(size_t)n * kH1 + k] = f2bf(W2[i]);
    }
}

// ---------------------------------------------------------------------------
// Kernel 1: one wave32 per batch row; lane == embedding dim (E == 32).
// Produces bf16 deep_emb row [emb(26*32) | title(32) | video(32)] and the
// scalar part  sum(fm_first) + sum(fm_second) + bias.
// ---------------------------------------------------------------------------
__global__ __launch_bounds__(256) void deepfm_embed(
    const int*   __restrict__ Xi,
    const float* __restrict__ Xv,
    const float* __restrict__ video_feature,
    const int*   __restrict__ title_feature,
    const float* __restrict__ title_value,
    const float* __restrict__ first_tables,
    const float* __restrict__ second_tables,
    const float* __restrict__ title_table,
    const float* __restrict__ video_W,
    const float* __restrict__ video_b,
    const float* __restrict__ bias,
    __bf16*      __restrict__ X,           // [B, 896] bf16
    float*       __restrict__ scalar_part) // [B]
{
    const int lane = threadIdx.x & 31;
    const int wave = threadIdx.x >> 5;
    const int row  = blockIdx.x * 8 + wave;
    const int e    = lane;

    const int*   xi   = Xi + row * kF;
    const float* xv   = Xv + row * kF;
    __bf16*      xrow = X + (size_t)row * kK1;

    // FM second-order accumulators along lane dim e
    float S = 0.f, Q = 0.f;
    for (int f = 0; f < kF; ++f) {
        int   idx = xi[f];
        float v   = second_tables[((size_t)f * kV + idx) * kE + e] * xv[f];
        xrow[f * kE + e] = f2bf(v);
        S += v;
        Q += v * v;
    }
    float part = 0.5f * (S * S - Q);             // lane e holds fm_second[e]
    if (lane < kF)                               // lane f holds fm_first[f]
        part += first_tables[(size_t)lane * kV + xi[lane]] * xv[lane];
    #pragma unroll
    for (int m = 16; m >= 1; m >>= 1)
        part += __shfl_xor(part, m, 32);

    // title embedding
    const int*   tf = title_feature + row * kL;
    const float* tv = title_value   + row * kL;
    float T = 0.f;
    for (int l = 0; l < kL; ++l)
        T += title_table[(size_t)tf[l] * kE + e] * tv[l];
    xrow[kF * kE + e] = f2bf(T);

    // video projection: video_feature(128) . video_W[e,:] + video_b[e]
    const float* vf = video_feature + (size_t)row * kDVF;
    const float* wr = video_W + (size_t)e * kDVF;
    float Vv = video_b[e];
    for (int k = 0; k < kDVF; ++k)
        Vv += vf[k] * wr[k];
    xrow[(kF + 1) * kE + e] = f2bf(Vv);

    if (lane == 0)
        scalar_part[row] = part + bias[0];
}

// ---------------------------------------------------------------------------
// 16x32 bf16 A-fragment per ISA 7.12.2:
//   lanes 0-15:  V0-3 = K 0-7,  V4-7 = K 16-23   (row M = lane)
//   lanes 16-31: V0-3 = K 8-15, V4-7 = K 24-31   (row M = lane-16)
// => two contiguous 16B chunks at (hi?8:0) and (hi?8:0)+16 halves.
// ---------------------------------------------------------------------------
__device__ __forceinline__ bf16x16 load_a_frag(const __bf16* arow, int k0) {
    bf16x8 lo = *(const bf16x8*)(arow + k0);
    bf16x8 hi = *(const bf16x8*)(arow + k0 + 16);
    return __builtin_shufflevector(lo, hi,
        0, 1, 2, 3, 4, 5, 6, 7, 8, 9, 10, 11, 12, 13, 14, 15);
}

// ---------------------------------------------------------------------------
// Kernel 2: H = relu(deep_emb @ W1 + b1)  M=16384, K=896, N=512 (bf16 WMMA)
// 16-row strip per workgroup; 8 waves x 4 n-tiles cover N=512.
// Fully unrolled k-loop (28 steps); sched-group pipeline keeps the 10 loads
// of step s+1 ahead of the 4 WMMAs of step s.
// ---------------------------------------------------------------------------
__global__ __launch_bounds__(256) void deepfm_gemm1(
    const __bf16* __restrict__ X,    // [B, 896]
    const __bf16* __restrict__ W1T,  // [512, 896] N-major
    const float*  __restrict__ bias1,
    __bf16*       __restrict__ H)    // [B, 512]
{
    const int  lane = threadIdx.x & 31;
    const int  wave = threadIdx.x >> 5;
    const bool hi   = lane >= 16;
    const int  m0   = blockIdx.x * 16;
    const int  n0   = wave * 64;
    const int  col  = lane & 15;

    const __bf16* arow = X + (size_t)(m0 + col) * kK1 + (hi ? 8 : 0);
    const __bf16* brow[4];
    #pragma unroll
    for (int j = 0; j < 4; ++j)
        brow[j] = W1T + (size_t)(n0 + j * 16 + col) * kK1 + (hi ? 8 : 0);

    f32x8 acc[4] = {};

    #pragma unroll
    for (int k0 = 0; k0 < kK1; k0 += 32) {
        bf16x16 a = load_a_frag(arow, k0);
        #pragma unroll
        for (int j = 0; j < 4; ++j) {
            bf16x16 b = load_a_frag(brow[j], k0);
            acc[j] = __builtin_amdgcn_wmma_f32_16x16x32_bf16(
                         false, a, false, b, (short)0, acc[j], false, false);
        }
    }

    // scheduler pipeline: loads of k-step s+1 before WMMAs of k-step s
    SCHED_GROUP(SG_VMEM_READ, 10, 0);            // loads for step 0
    #pragma unroll
    for (int s = 0; s < kK1 / 32 - 1; ++s) {     // 27 steady-state stages
        SCHED_GROUP(SG_VMEM_READ, 10, 0);        // loads for step s+1
        SCHED_GROUP(SG_WMMA, 4, 0);              // WMMAs for step s
    }
    SCHED_GROUP(SG_WMMA, 4, 0);                  // WMMAs for last step

    // epilogue: +bias, ReLU, store bf16.  C layout: VGPR r -> M = r + (hi?8:0)
    #pragma unroll
    for (int j = 0; j < 4; ++j) {
        int   n  = n0 + j * 16 + col;
        float bv = bias1[n];
        #pragma unroll
        for (int r = 0; r < 8; ++r) {
            int   m = m0 + r + (hi ? 8 : 0);
            float v = acc[j][r] + bv;
            v = v > 0.f ? v : 0.f;
            H[(size_t)m * kH1 + n] = f2bf(v);
        }
    }
}

// ---------------------------------------------------------------------------
// Kernel 3: x = relu(H @ W2 + b2); out[b] = scalar_part[b] + sum_n x[b,n]
// M=16384, K=512, N=256; 8 waves x 2 n-tiles; fully unrolled (16 steps) with
// sched-group pipeline; row-sum via lane butterfly + LDS atomics.
// ---------------------------------------------------------------------------
__global__ __launch_bounds__(256) void deepfm_gemm2(
    const __bf16* __restrict__ H,    // [B, 512]
    const __bf16* __restrict__ W2T,  // [256, 512] N-major
    const float*  __restrict__ bias2,
    const float*  __restrict__ scalar_part,
    float*        __restrict__ out)
{
    __shared__ float rowsum[16];
    const int  lane = threadIdx.x & 31;
    const int  wave = threadIdx.x >> 5;
    const bool hi   = lane >= 16;
    const int  m0   = blockIdx.x * 16;
    const int  n0   = wave * 32;
    const int  col  = lane & 15;

    if (threadIdx.x < 16) rowsum[threadIdx.x] = 0.f;
    __syncthreads();

    const __bf16* arow = H + (size_t)(m0 + col) * kH1 + (hi ? 8 : 0);
    const __bf16* brow[2];
    #pragma unroll
    for (int j = 0; j < 2; ++j)
        brow[j] = W2T + (size_t)(n0 + j * 16 + col) * kH1 + (hi ? 8 : 0);

    f32x8 acc[2] = {};

    #pragma unroll
    for (int k0 = 0; k0 < kH1; k0 += 32) {
        bf16x16 a = load_a_frag(arow, k0);
        #pragma unroll
        for (int j = 0; j < 2; ++j) {
            bf16x16 b = load_a_frag(brow[j], k0);
            acc[j] = __builtin_amdgcn_wmma_f32_16x16x32_bf16(
                         false, a, false, b, (short)0, acc[j], false, false);
        }
    }

    SCHED_GROUP(SG_VMEM_READ, 6, 0);             // loads for step 0
    #pragma unroll
    for (int s = 0; s < kH1 / 32 - 1; ++s) {     // 15 steady-state stages
        SCHED_GROUP(SG_VMEM_READ, 6, 0);         // loads for step s+1
        SCHED_GROUP(SG_WMMA, 2, 0);              // WMMAs for step s
    }
    SCHED_GROUP(SG_WMMA, 2, 0);                  // WMMAs for last step

    float bv0 = bias2[n0 + col];
    float bv1 = bias2[n0 + 16 + col];
    #pragma unroll
    for (int r = 0; r < 8; ++r) {
        float v0 = acc[0][r] + bv0; v0 = v0 > 0.f ? v0 : 0.f;
        float v1 = acc[1][r] + bv1; v1 = v1 > 0.f ? v1 : 0.f;
        float s  = v0 + v1;              // same M, two different N columns
        s += __shfl_xor(s, 1, 32);       // sum over the 16 N-lanes of my half
        s += __shfl_xor(s, 2, 32);
        s += __shfl_xor(s, 4, 32);
        s += __shfl_xor(s, 8, 32);
        if (col == 0)
            atomicAdd(&rowsum[r + (hi ? 8 : 0)], s);   // ds_add_f32
    }
    __syncthreads();
    if (threadIdx.x < 16) {
        int m = m0 + threadIdx.x;
        out[m] = scalar_part[m] + rowsum[threadIdx.x];
    }
}

// ---------------------------------------------------------------------------
extern "C" void kernel_launch(void* const* d_in, const int* in_sizes, int n_in,
                              void* d_out, int out_size, void* d_ws, size_t ws_size,
                              hipStream_t stream)
{
    (void)in_sizes; (void)n_in; (void)out_size; (void)ws_size;

    const int*   Xi    = (const int*)  d_in[0];
    const float* Xv    = (const float*)d_in[1];
    const float* vfeat = (const float*)d_in[2];
    const int*   tfeat = (const int*)  d_in[3];
    const float* tval  = (const float*)d_in[4];
    const float* ftab  = (const float*)d_in[5];
    const float* stab  = (const float*)d_in[6];
    const float* ttab  = (const float*)d_in[7];
    const float* vW    = (const float*)d_in[8];
    const float* vb    = (const float*)d_in[9];
    const float* W1    = (const float*)d_in[10];
    const float* b1    = (const float*)d_in[11];
    const float* W2    = (const float*)d_in[12];
    const float* b2    = (const float*)d_in[13];
    const float* bias  = (const float*)d_in[14];
    float*       out   = (float*)d_out;

    // workspace carve-up (~47.4 MB), 256B-aligned sections
    char*  ws  = (char*)d_ws;
    size_t off = 0;
    __bf16* X    = (__bf16*)(ws + off); off += (size_t)kB  * kK1 * 2;  // deep_emb
    __bf16* Hbuf = (__bf16*)(ws + off); off += (size_t)kB  * kH1 * 2;  // hidden1
    __bf16* W1T  = (__bf16*)(ws + off); off += (size_t)kH1 * kK1 * 2;
    __bf16* W2T  = (__bf16*)(ws + off); off += (size_t)kH2 * kH1 * 2;
    float*  scal = (float*)(ws + off);  off += (size_t)kB * sizeof(float);

    const int prepN = kK1 * kH1 + kH1 * kH2;
    hipLaunchKernelGGL(deepfm_prep, dim3((prepN + 255) / 256), dim3(256), 0, stream,
                       W1, W2, W1T, W2T);
    hipLaunchKernelGGL(deepfm_embed, dim3(kB / 8), dim3(256), 0, stream,
                       Xi, Xv, vfeat, tfeat, tval, ftab, stab, ttab, vW, vb, bias,
                       X, scal);
    hipLaunchKernelGGL(deepfm_gemm1, dim3(kB / 16), dim3(256), 0, stream,
                       X, W1T, b1, Hbuf);
    hipLaunchKernelGGL(deepfm_gemm2, dim3(kB / 16), dim3(256), 0, stream,
                       Hbuf, W2T, b2, scal, out);
}
